// DTM_15960098472532
// MI455X (gfx1250) — matrix operand order repaired and checked by reading, not compiled, not run on hardware
//
#include <hip/hip_runtime.h>
#include <hip/hip_bf16.h>

// ---------------------------------------------------------------------------
// Problem constants (from reference):
//   B=16, C=96, H=W=128, KS=ST=2 -> Ho=Wo=64, KK=4, O=2C=192
//   M = B*Ho*Wo = 65536 rows, K = C*KK = 384, N = 192
// ---------------------------------------------------------------------------
#define Bn    16
#define Cc    96
#define Hh    128
#define Ww    128
#define HoW   64          // Ho = Wo = 64
#define SPAT  4096        // Ho*Wo
#define Oc    192
#define Mrows 65536       // B * SPAT
#define Kdim  384         // C * KK
#define Ndim  192

typedef __bf16 bf16_t;
typedef __attribute__((ext_vector_type(16))) __bf16 v16bf;
typedef __attribute__((ext_vector_type(8)))  __bf16 v8bf;
typedef __attribute__((ext_vector_type(8)))  float  v8f;

// ---------------------------------------------------------------------------
// Kernel 1: stride-2 2x2 offset conv.  One block per (b, ho, wo).
// Loads the 2x2xC patch to LDS, then 8 waves each reduce one output channel.
// ---------------------------------------------------------------------------
__global__ void __launch_bounds__(256)
dtm_offset_conv(const float* __restrict__ x, const float* __restrict__ w_off,
                const float* __restrict__ b_off, float* __restrict__ off) {
  __shared__ float patch[Kdim];
  const int m  = blockIdx.x;
  const int b  = m >> 12;
  const int sp = m & (SPAT - 1);
  const int ho = sp >> 6, wo = sp & 63;
  const float* xb = x + (size_t)b * (Hh * Ww * Cc);
  for (int t = threadIdx.x; t < Kdim; t += 256) {
    const int kidx = t / Cc, c = t - kidx * Cc;
    const int iy = ho * 2 + (kidx >> 1);
    const int ix = wo * 2 + (kidx & 1);
    patch[t] = xb[((iy << 7) + ix) * Cc + c];
  }
  __syncthreads();
  const int ch   = threadIdx.x >> 5;   // wave id == output channel 0..7
  const int lane = threadIdx.x & 31;
  float partial = 0.f;
  for (int j = lane; j < Kdim; j += 32) {
    const int kidx = j / Cc, c = j - kidx * Cc;
    partial += patch[j] * w_off[ch * Kdim + c * 4 + kidx];  // (ch,c,ki,kj) flat
  }
  for (int o = 16; o > 0; o >>= 1) partial += __shfl_xor(partial, o, 32);
  if (lane == 0) off[((b * 8 + ch) << 12) + sp] = partial + b_off[ch];
}

// ---------------------------------------------------------------------------
// Kernel 2: cast w_dconv (192*96*4 f32, flat layout == Wt[n][k]) to bf16.
// ---------------------------------------------------------------------------
__global__ void __launch_bounds__(256)
dtm_cast_w(const float* __restrict__ w, bf16_t* __restrict__ wt) {
  const int i = blockIdx.x * 256 + threadIdx.x;
  if (i < Ndim * Kdim) wt[i] = (bf16_t)w[i];
}

__device__ __forceinline__ float dtm_fetch(const float* __restrict__ xb,
                                           int iy, int ix, int c) {
  if ((unsigned)iy < (unsigned)Hh && (unsigned)ix < (unsigned)Ww)
    return xb[((iy << 7) + ix) * Cc + c];
  return 0.f;
}

// ---------------------------------------------------------------------------
// Kernel 3: bilinear sampling -> bf16 A matrix [M][K], K index = c*4 + kk
// One block per spatial row m; 384 threads: t = kk*96 + c (c contiguous ->
// coalesced channel-major gathers from x).
// ---------------------------------------------------------------------------
__global__ void __launch_bounds__(384)
dtm_sample(const float* __restrict__ x, const float* __restrict__ off,
           bf16_t* __restrict__ A) {
  const int m  = blockIdx.x;
  const int t  = threadIdx.x;
  const int kk = t / Cc;
  const int c  = t - kk * Cc;
  const int b  = m >> 12;
  const int sp = m & (SPAT - 1);
  const int ho = sp >> 6, wo = sp & 63;

  const float dy = off[((b * 8 + kk * 2 + 0) << 12) + sp];
  const float dx = off[((b * 8 + kk * 2 + 1) << 12) + sp];
  const float py = (float)(ho * 2 + (kk >> 1)) + dy;
  const float px = (float)(wo * 2 + (kk & 1)) + dx;
  const float fy0 = floorf(py), fx0 = floorf(px);
  const float wy1 = py - fy0, wx1 = px - fx0;
  const float wy0 = 1.f - wy1, wx0 = 1.f - wx1;
  const int iy0 = (int)fy0, ix0 = (int)fx0;

  const float* xb = x + (size_t)b * (Hh * Ww * Cc);
  const float v00 = dtm_fetch(xb, iy0,     ix0,     c);
  const float v01 = dtm_fetch(xb, iy0,     ix0 + 1, c);
  const float v10 = dtm_fetch(xb, iy0 + 1, ix0,     c);
  const float v11 = dtm_fetch(xb, iy0 + 1, ix0 + 1, c);
  const float val = v00 * (wy0 * wx0) + v01 * (wy0 * wx1)
                  + v10 * (wy1 * wx0) + v11 * (wy1 * wx1);
  A[(size_t)m * Kdim + c * 4 + kk] = (bf16_t)val;
}

// ---------------------------------------------------------------------------
// Kernel 4: WMMA GEMM  Y[M][N] = A[M][K](bf16) * Wt[N][K]^T(bf16), f32 accum.
// One wave per 16x16 output tile. K loop: 12 steps of v_wmma_f32_16x16x32_bf16.
//   A frag (16x32): lane l -> row m=l&15, h=l>>4; elems e at K=h*16+e
//                   -> two 16B loads.
//   B frag (32x16): lane l -> col n=l&15, h=l>>4; elems e at K=h*16+e
//                   -> one contiguous 32B load from row-major Wt[n][.].
//   D (16x16 f32):  reg r -> row r+8h, lane -> col.
// ---------------------------------------------------------------------------
__global__ void __launch_bounds__(256)
dtm_wmma_gemm(const bf16_t* __restrict__ A, const bf16_t* __restrict__ Wt,
              float* __restrict__ Y) {
  const int wid   = (blockIdx.x * 256 + threadIdx.x) >> 5;
  const int lane  = threadIdx.x & 31;
  const int mtile = wid / (Ndim / 16);
  const int ntile = wid - mtile * (Ndim / 16);
  const int mn    = lane & 15;     // row within A-tile / col within B-tile
  const int h     = lane >> 4;

  const bf16_t* Arow = A  + (size_t)(mtile * 16 + mn) * Kdim;
  const bf16_t* Brow = Wt + (size_t)(ntile * 16 + mn) * Kdim;

  v8f acc = {};
#pragma unroll
  for (int s = 0; s < Kdim / 32; ++s) {
    const int base = s * 32;
    union { v16bf v; struct { v8bf lo; v8bf hi; } p; } a;
    a.p.lo = *(const v8bf*)(Arow + base + h * 16);       // K = h*16 + 0..7
    a.p.hi = *(const v8bf*)(Arow + base + h * 16 + 8);   // K = h*16 + 8..15
    const v16bf bv = *(const v16bf*)(Brow + base + h * 16); // K = h*16 + 0..15
    acc = __builtin_amdgcn_wmma_f32_16x16x32_bf16(
        /*neg_a=*/false, a.v, /*neg_b=*/false, bv,
        /*c_mod=*/(short)0, acc, /*reuse_a=*/false, /*reuse_b=*/false);
  }

  float* yout = Y + (size_t)(mtile * 16) * Ndim + ntile * 16 + mn;
#pragma unroll
  for (int r = 0; r < 8; ++r)
    yout[(size_t)(r + 8 * h) * Ndim] = acc[r];
}

// ---------------------------------------------------------------------------
// Kernels 5-7: BN statistics (two-stage atomic reduction), scale/shift
// finalize, fused BN + exact GELU writeout (output already (B, L', O)).
// stats layout: [0..191]=sum, [192..383]=sumsq, [384..575]=scale, [576..767]=shift
// ---------------------------------------------------------------------------
__global__ void dtm_zero_stats(float* __restrict__ stats) {
  if (threadIdx.x < 384) stats[threadIdx.x] = 0.f;
}

__global__ void __launch_bounds__(192)
dtm_stats(const float* __restrict__ Y, float* __restrict__ stats) {
  const int n = threadIdx.x;
  const size_t r0 = (size_t)blockIdx.x * 256;
  float s = 0.f, s2 = 0.f;
  for (int r = 0; r < 256; ++r) {
    const float v = Y[(r0 + r) * Ndim + n];
    s += v; s2 += v * v;
  }
  atomicAdd(&stats[n], s);
  atomicAdd(&stats[Ndim + n], s2);
}

__global__ void __launch_bounds__(192)
dtm_finalize_stats(float* __restrict__ stats, const float* __restrict__ gamma,
                   const float* __restrict__ beta) {
  const int n = threadIdx.x;
  const float inv = 1.f / (float)Mrows;
  const float mean = stats[n] * inv;
  const float var  = stats[Ndim + n] * inv - mean * mean;
  const float sc   = gamma[n] * rsqrtf(var + 1e-5f);
  stats[2 * Ndim + n] = sc;
  stats[3 * Ndim + n] = beta[n] - mean * sc;
}

__global__ void __launch_bounds__(256)
dtm_bn_gelu(const float* __restrict__ Y, const float* __restrict__ stats,
            float* __restrict__ out) {
  const int i = blockIdx.x * 256 + threadIdx.x;
  const int n = i % Ndim;
  const float t = Y[i] * stats[2 * Ndim + n] + stats[3 * Ndim + n];
  out[i] = 0.5f * t * (1.f + erff(t * 0.70710678118654752440f));
}

// ---------------------------------------------------------------------------
// Host launcher
// ---------------------------------------------------------------------------
extern "C" void kernel_launch(void* const* d_in, const int* in_sizes, int n_in,
                              void* d_out, int out_size, void* d_ws, size_t ws_size,
                              hipStream_t stream) {
  (void)in_sizes; (void)n_in; (void)out_size; (void)ws_size;
  const float* x        = (const float*)d_in[0];
  const float* w_offset = (const float*)d_in[1];
  const float* b_offset = (const float*)d_in[2];
  const float* w_dconv  = (const float*)d_in[3];
  const float* bn_gamma = (const float*)d_in[4];
  const float* bn_beta  = (const float*)d_in[5];
  float* out = (float*)d_out;

  // Workspace layout (all regions 256B aligned):
  //   off   : B*8*SPAT f32            =  2,097,152 B
  //   Wt    : N*K bf16                =    147,456 B
  //   A     : M*K bf16                = 50,331,648 B
  //   Y     : M*N f32                 = 50,331,648 B
  //   stats : 4*192 f32               =      3,072 B   (~103 MB total)
  char* ws = (char*)d_ws;
  float*  off   = (float*)(ws + 0);
  bf16_t* Wt    = (bf16_t*)(ws + 2097152);
  bf16_t* A     = (bf16_t*)(ws + 2097152 + 147456);
  float*  Y     = (float*)(ws + 2097152 + 147456 + 50331648);
  float*  stats = (float*)(ws + 2097152 + 147456 + 50331648 + 50331648);

  dtm_offset_conv<<<Mrows, 256, 0, stream>>>(x, w_offset, b_offset, off);
  dtm_cast_w<<<(Ndim * Kdim + 255) / 256, 256, 0, stream>>>(w_dconv, Wt);
  dtm_sample<<<Mrows, 384, 0, stream>>>(x, off, A);
  dtm_wmma_gemm<<<(Mrows / 16) * (Ndim / 16) / 8, 256, 0, stream>>>(A, Wt, Y);
  dtm_zero_stats<<<1, 384, 0, stream>>>(stats);
  dtm_stats<<<Mrows / 256, 192, 0, stream>>>(Y, stats);
  dtm_finalize_stats<<<1, 192, 0, stream>>>(stats, bn_gamma, bn_beta);
  dtm_bn_gelu<<<(Mrows * Ndim) / 256, 256, 0, stream>>>(Y, stats, out);
}
